// MultiheadAttention_12232066859501
// MI455X (gfx1250) — matrix operand order repaired and checked by reading, not compiled, not run on hardware
//
#include <hip/hip_runtime.h>
#include <hip/hip_bf16.h>

// ---------------------------------------------------------------------------
// T5-style multihead attention for MI455X (gfx1250, wave32, WMMA).
// T=2048, B=2, E=768, H=12, D=64. bf16 WMMA with fp32 accumulation.
// - WMMA fragment gathers are ds_load_b128 pairs, except Q (global_load_b128 pairs).
// - All pure-copy tile staging uses global_load_async_to_lds_b128 (ASYNCcnt).
// - T5 relative bias precomputed per head as a bias-by-distance table; the
//   flash loop reads a 128-entry LDS window instead of doing log math.
// ---------------------------------------------------------------------------

#define EMBED 768
#define HEADS 12
#define HDIM  64
#define TLEN  2048
#define BATCH 2
#define NBUCK 32
#define TBIAS 4352      // bias-by-distance table entries per head
#define TBOFF 2112      // table index = rel + TBOFF

typedef __attribute__((ext_vector_type(16))) __bf16          v16bf;
typedef __attribute__((ext_vector_type(8)))  __bf16          v8bf;
typedef __attribute__((ext_vector_type(8)))  float           v8f;
typedef __attribute__((ext_vector_type(8)))  unsigned short  us8;

__device__ __forceinline__ unsigned short f2bf(float f) {
  return __builtin_bit_cast(unsigned short, (__bf16)f);   // native cvt, RNE
}
__device__ __forceinline__ v8f zero8() {
  v8f z;
#pragma unroll
  for (int i = 0; i < 8; ++i) z[i] = 0.0f;
  return z;
}
__device__ __forceinline__ v8bf ld8(const unsigned short* p) {
  return *(const v8bf*)p;                   // -> ds_load_b128 / global_load_b128
}
__device__ __forceinline__ v16bf cat16(v8bf lo, v8bf hi) {
  return __builtin_shufflevector(lo, hi, 0, 1, 2, 3, 4, 5, 6, 7, 8, 9, 10, 11,
                                 12, 13, 14, 15);
}
__device__ __forceinline__ v8f wmma_bf16(v16bf a, v16bf b, v8f c) {
  return __builtin_amdgcn_wmma_f32_16x16x32_bf16(false, a, false, b,
                                                 (short)0, c, false, false);
}
// Async DMA: one 16B chunk global -> LDS, tracked by ASYNCcnt.
__device__ __forceinline__ void async_b128(unsigned lds_addr, unsigned goff,
                                           const void* sbase) {
  asm volatile("global_load_async_to_lds_b128 %0, %1, %2"
               :: "v"(lds_addr), "v"(goff),
                  "s"((unsigned long long)sbase) : "memory");
}
__device__ __forceinline__ void wait_async0() {
  asm volatile("s_wait_asynccnt 0x0" ::: "memory");
}

// ---------------------------------------------------------------------------
// Kernel 1: fp32 -> bf16 weight conversion WITH transpose:
//   W [in=768][out=768]  ->  Wt [out=768][in=768]   (B-operand K-contiguous)
// ---------------------------------------------------------------------------
__global__ void k_wcvt_t(const float* __restrict__ s,
                         unsigned short* __restrict__ d) {
  int i = blockIdx.x * 256 + threadIdx.x;
  if (i < EMBED * EMBED) {
    int r = i / EMBED, c = i % EMBED;       // coalesced read, scattered write
    d[(size_t)c * EMBED + r] = f2bf(s[i]);
  }
}

// ---------------------------------------------------------------------------
// Kernel 1b: per-head T5 bias-by-distance table.
//   tab[h][idx] = rel_bias[bucket(idx - TBOFF)][h]   (log math done ONCE here)
// ---------------------------------------------------------------------------
__global__ void k_biasrel(const float* __restrict__ rel_bias,
                          float* __restrict__ tab) {
  int i = blockIdx.x * 256 + threadIdx.x;
  if (i < HEADS * TBIAS) {
    int h = i / TBIAS, idx = i % TBIAS;
    int rel = idx - TBOFF;
    int bucket = (rel > 0) ? 16 : 0;
    int rp = (rel < 0) ? -rel : rel;
    int add;
    if (rp < 8) add = rp;
    else {
      // 8 + floor(log(rp/8)/log(16) * 8), clamped to 15
      int lg = 8 + (int)(__logf((float)rp * 0.125f) * 2.8853902f);
      add = (lg < 15) ? lg : 15;
    }
    tab[i] = rel_bias[(bucket + add) * HEADS + h];
  }
}

// ---------------------------------------------------------------------------
// Kernel 2: fused QKV projection GEMM.
//  X [4096 x 768] fp32 (row = t*B + b)  x  Wt^T  (+bias)*scale
//  q,k -> bf16 [B,H,T,D];  v -> bf16 [B,H,D,T] (transposed for P*V B-frags)
//  grid = (64, 6, 3), block = 256 (8 waves); wave tile 16x64, K staged 32.
// ---------------------------------------------------------------------------
__global__ __launch_bounds__(256) void k_qkv_gemm(
    const float* __restrict__ xq, const float* __restrict__ xk,
    const float* __restrict__ xv,
    const unsigned short* __restrict__ wq, const unsigned short* __restrict__ wk,
    const unsigned short* __restrict__ wv,
    const float* __restrict__ bq, const float* __restrict__ bk,
    const float* __restrict__ bv,
    unsigned short* __restrict__ qh, unsigned short* __restrict__ kh,
    unsigned short* __restrict__ vh) {
  const int z = blockIdx.z;
  const float*          X    = (z == 0) ? xq : (z == 1) ? xk : xv;
  const unsigned short* Wt   = (z == 0) ? wq : (z == 1) ? wk : wv;
  const float*          bias = (z == 0) ? bq : (z == 1) ? bk : bv;
  unsigned short*       O    = (z == 0) ? qh : (z == 1) ? kh : vh;
  const float scale = (z == 0) ? 0.125f : 1.0f;   // 1/sqrt(64) for q

  const int m0 = blockIdx.x * 64;
  const int n0 = blockIdx.y * 128;
  const int tid = threadIdx.x;
  const int wave = tid >> 5, lane = tid & 31;
  const int lane16 = lane & 15, half = lane >> 4;
  const int wm = wave >> 1;   // 0..3: 16-row group
  const int wn = wave & 1;    // 0..1: 64-col group

  __shared__ __align__(16) unsigned short As[64][40];    // [row][k], padded
  __shared__ __align__(16) unsigned short Bt[128][40];   // [col][k], padded

  v8f acc[4];
#pragma unroll
  for (int i = 0; i < 4; ++i) acc[i] = zero8();

  for (int k0 = 0; k0 < EMBED; k0 += 32) {
    // B tile: async DMA (pure bf16 copy from transposed weights)
#pragma unroll
    for (int cc = 0; cc < 2; ++cc) {
      int idx = tid + cc * 256;        // 0..511
      int c = idx >> 2, r = (idx & 3) * 8;
      async_b128((unsigned)(uintptr_t)&Bt[c][r],
                 (unsigned)(((n0 + c) * EMBED + k0 + r) * 2), Wt);
    }
    {   // A tile: fp32->bf16 conversion through VGPRs, 128-bit LDS store
      int r = tid >> 2, c = (tid & 3) * 8;
      const float* xp = &X[(size_t)(m0 + r) * EMBED + k0 + c];
      us8 pk;
#pragma unroll
      for (int u = 0; u < 8; ++u) pk[u] = f2bf(xp[u]);
      *(us8*)&As[r][c] = pk;
    }
    wait_async0();
    __syncthreads();

    const int row = wm * 16 + lane16;
    v16bf a = cat16(ld8(&As[row][half * 8]), ld8(&As[row][16 + half * 8]));
#pragma unroll
    for (int ns = 0; ns < 4; ++ns) {
      int col = wn * 64 + ns * 16 + lane16;
      v16bf bmat = cat16(ld8(&Bt[col][half * 8]), ld8(&Bt[col][16 + half * 8]));
      acc[ns] = wmma_bf16(a, bmat, acc[ns]);
    }
    __syncthreads();
  }

  // epilogue: (acc + bias) * scale
#pragma unroll
  for (int ns = 0; ns < 4; ++ns) {
    int col = n0 + wn * 64 + ns * 16 + lane16;
    int h = col >> 6, d = col & 63;
    float bb = bias[col];
#pragma unroll
    for (int j = 0; j < 8; ++j) {
      int row = m0 + wm * 16 + half * 8 + j;   // = t*B + b
      int t = row >> 1, b = row & 1;
      unsigned short v = f2bf((acc[ns][j] + bb) * scale);
      if (z == 2)   // V transposed: [B,H,D,T]
        O[((size_t)(b * HEADS + h) * HDIM + d) * TLEN + t] = v;
      else          // Q,K: [B,H,T,D]
        O[((size_t)(b * HEADS + h) * TLEN + t) * HDIM + d] = v;
    }
  }
}

// ---------------------------------------------------------------------------
// Kernel 3: flash attention.  grid = (T/64, B*H), block = 128 (4 waves).
// K/V tiles staged via global_load_async_to_lds_b128; bias from per-head
// distance table via a 128-float LDS window; online softmax in fp32.
// ---------------------------------------------------------------------------
__global__ __launch_bounds__(128) void k_attn(
    const unsigned short* __restrict__ qh, const unsigned short* __restrict__ kh,
    const unsigned short* __restrict__ vh, const unsigned char* __restrict__ kpm,
    const float* __restrict__ biasrel, unsigned short* __restrict__ attn) {
  const int bh = blockIdx.y;             // b*HEADS + h
  const int b = bh / HEADS, h = bh % HEADS;
  const int tid = threadIdx.x;
  const int wave = tid >> 5, lane = tid & 31;
  const int lane16 = lane & 15, half = lane >> 4;
  const int t_blk = blockIdx.x * 64;
  const int t_wave = t_blk + wave * 16;

  __shared__ __align__(16) unsigned short Ks[64][72];      // [s][d], padded
  __shared__ __align__(16) unsigned short Vt[64][72];      // [d][s], padded
  __shared__ __align__(16) unsigned short Ps[4][16][72];   // per-wave P tiles
  __shared__ float win[128];                               // bias window
  __shared__ float smask[64];

  const unsigned short* qbase  = qh + (size_t)bh * TLEN * HDIM;
  const unsigned short* kbase  = kh + (size_t)bh * TLEN * HDIM;
  const unsigned short* vbaseT = vh + (size_t)bh * HDIM * TLEN;  // [d][t]
  const float*          btab   = biasrel + (size_t)h * TBIAS;

  // Q fragments (A-layout): two b128 loads per K=32 chunk
  v16bf qa[2];
  {
    const unsigned short* qr = qbase + (size_t)(t_wave + lane16) * HDIM;
#pragma unroll
    for (int c = 0; c < 2; ++c)
      qa[c] = cat16(ld8(qr + c * 32 + half * 8), ld8(qr + c * 32 + 16 + half * 8));
  }

  v8f acc[4];
  float m[8], l[8];
#pragma unroll
  for (int i = 0; i < 4; ++i) acc[i] = zero8();
#pragma unroll
  for (int j = 0; j < 8; ++j) { m[j] = -3.0e38f; l[j] = 0.0f; }

  for (int s0 = 0; s0 < TLEN; s0 += 64) {
    // Async DMA of K and V tiles into LDS (tracked by ASYNCcnt)
#pragma unroll
    for (int cc = 0; cc < 4; ++cc) {
      int idx = tid + cc * 128;          // 0..511 chunks of 8 bf16
      int r = idx >> 3, c8 = (idx & 7) * 8;
      async_b128((unsigned)(uintptr_t)&Ks[r][c8],
                 (unsigned)(((s0 + r) * HDIM + c8) * 2), kbase);
      async_b128((unsigned)(uintptr_t)&Vt[r][c8],
                 (unsigned)((r * TLEN + s0 + c8) * 2), vbaseT);
    }
    // bias window: rel in [s0-t_blk-63, s0-t_blk+64]
    if (tid < 128) win[tid] = btab[(s0 - t_blk - 63 + TBOFF) + tid];
    if (tid < 64)
      smask[tid] = kpm[(size_t)b * TLEN + s0 + tid] ? -1.0e30f : 0.0f;
    if (s0 + 64 < TLEN) {  // speculative prefetch of next tiles
      __builtin_prefetch(kbase + (size_t)(s0 + 64) * HDIM + lane * 64, 0, 0);
      __builtin_prefetch(vbaseT + (size_t)lane * TLEN + s0 + 64, 0, 0);
    }
    wait_async0();
    __syncthreads();

    // scores = Q * K^T : B-frag = two 16-byte pieces of a K row (table order)
    v8f st[4];
#pragma unroll
    for (int ns = 0; ns < 4; ++ns) {
      st[ns] = zero8();
      int sl = ns * 16 + lane16;
#pragma unroll
      for (int c = 0; c < 2; ++c) {
        v16bf kb = cat16(ld8(&Ks[sl][c * 32 + half * 8]),
                         ld8(&Ks[sl][c * 32 + 16 + half * 8]));
        st[ns] = wmma_bf16(qa[c], kb, st[ns]);
      }
    }

    // bias (from LDS window, index = rel - rel0 = basei - j) + padding mask
#pragma unroll
    for (int ns = 0; ns < 4; ++ns) {
      float mk = smask[ns * 16 + lane16];
      int basei = ns * 16 + lane16 + 63 - wave * 16 - half * 8;
#pragma unroll
      for (int j = 0; j < 8; ++j) st[ns][j] += win[basei - j] + mk;
    }

    // online softmax: row reductions across the 16 lanes owning a row
    float nm[8], rs[8];
#pragma unroll
    for (int j = 0; j < 8; ++j) {
      float mx = fmaxf(fmaxf(st[0][j], st[1][j]), fmaxf(st[2][j], st[3][j]));
#pragma unroll
      for (int xm = 1; xm < 16; xm <<= 1) mx = fmaxf(mx, __shfl_xor(mx, xm, 32));
      nm[j] = fmaxf(m[j], mx);
    }
#pragma unroll
    for (int j = 0; j < 8; ++j) {
      float sum = 0.0f;
#pragma unroll
      for (int ns = 0; ns < 4; ++ns) {
        float p = __expf(st[ns][j] - nm[j]);
        st[ns][j] = p;
        sum += p;
      }
#pragma unroll
      for (int xm = 1; xm < 16; xm <<= 1) sum += __shfl_xor(sum, xm, 32);
      rs[j] = sum;
    }
#pragma unroll
    for (int j = 0; j < 8; ++j) {
      float sc = __expf(m[j] - nm[j]);
      l[j] = l[j] * sc + rs[j];
      m[j] = nm[j];
#pragma unroll
      for (int ns = 0; ns < 4; ++ns) acc[ns][j] *= sc;
    }

    // C-layout -> A-layout transpose of P via per-wave LDS region
#pragma unroll
    for (int ns = 0; ns < 4; ++ns)
#pragma unroll
      for (int j = 0; j < 8; ++j)
        Ps[wave][half * 8 + j][ns * 16 + lane16] = f2bf(st[ns][j]);
    asm volatile("s_wait_dscnt 0x0" ::: "memory");  // intra-wave LDS RAW

    v16bf pa[2];
#pragma unroll
    for (int kc = 0; kc < 2; ++kc)
      pa[kc] = cat16(ld8(&Ps[wave][lane16][kc * 32 + half * 8]),
                     ld8(&Ps[wave][lane16][kc * 32 + 16 + half * 8]));

    // acc += P(16x64) * V(64x64); V B-frag = two 16-byte pieces of Vt row (table order)
#pragma unroll
    for (int ns = 0; ns < 4; ++ns) {
      int dcol = ns * 16 + lane16;
#pragma unroll
      for (int kc = 0; kc < 2; ++kc) {
        v16bf vb = cat16(ld8(&Vt[dcol][kc * 32 + half * 8]),
                         ld8(&Vt[dcol][kc * 32 + 16 + half * 8]));
        acc[ns] = wmma_bf16(pa[kc], vb, acc[ns]);
      }
    }
    __syncthreads();
  }

  // normalize and write bf16 [T,B,E]
#pragma unroll
  for (int j = 0; j < 8; ++j) {
    int t = t_wave + half * 8 + j;
    float inv = 1.0f / l[j];
#pragma unroll
    for (int ns = 0; ns < 4; ++ns) {
      int d = ns * 16 + lane16;
      attn[((size_t)(t * BATCH + b)) * EMBED + h * HDIM + d] =
          f2bf(acc[ns][j] * inv);
    }
  }
}

// ---------------------------------------------------------------------------
// Kernel 4: output projection.  attn bf16 [4096 x 768] x Wo (transposed) + bo
// Both tiles staged with async DMA.
// ---------------------------------------------------------------------------
__global__ __launch_bounds__(256) void k_out_gemm(
    const unsigned short* __restrict__ A, const unsigned short* __restrict__ Wt,
    const float* __restrict__ bias, float* __restrict__ out) {
  const int m0 = blockIdx.x * 64;
  const int n0 = blockIdx.y * 128;
  const int tid = threadIdx.x;
  const int wave = tid >> 5, lane = tid & 31;
  const int lane16 = lane & 15, half = lane >> 4;
  const int wm = wave >> 1, wn = wave & 1;

  __shared__ __align__(16) unsigned short As[64][40];
  __shared__ __align__(16) unsigned short Bt[128][40];

  v8f acc[4];
#pragma unroll
  for (int i = 0; i < 4; ++i) acc[i] = zero8();

  for (int k0 = 0; k0 < EMBED; k0 += 32) {
    {
      int r = tid >> 2, c = (tid & 3) * 8;
      async_b128((unsigned)(uintptr_t)&As[r][c],
                 (unsigned)(((m0 + r) * EMBED + k0 + c) * 2), A);
    }
#pragma unroll
    for (int cc = 0; cc < 2; ++cc) {
      int idx = tid + cc * 256;
      int c = idx >> 2, r = (idx & 3) * 8;
      async_b128((unsigned)(uintptr_t)&Bt[c][r],
                 (unsigned)(((n0 + c) * EMBED + k0 + r) * 2), Wt);
    }
    wait_async0();
    __syncthreads();

    const int row = wm * 16 + lane16;
    v16bf a = cat16(ld8(&As[row][half * 8]), ld8(&As[row][16 + half * 8]));
#pragma unroll
    for (int ns = 0; ns < 4; ++ns) {
      int col = wn * 64 + ns * 16 + lane16;
      v16bf bmat = cat16(ld8(&Bt[col][half * 8]), ld8(&Bt[col][16 + half * 8]));
      acc[ns] = wmma_bf16(a, bmat, acc[ns]);
    }
    __syncthreads();
  }

#pragma unroll
  for (int ns = 0; ns < 4; ++ns) {
    int col = n0 + wn * 64 + ns * 16 + lane16;
    float bb = bias[col];
#pragma unroll
    for (int j = 0; j < 8; ++j) {
      int row = m0 + wm * 16 + half * 8 + j;
      out[(size_t)row * EMBED + col] = acc[ns][j] + bb;
    }
  }
}

// ---------------------------------------------------------------------------
extern "C" void kernel_launch(void* const* d_in, const int* in_sizes, int n_in,
                              void* d_out, int out_size, void* d_ws,
                              size_t ws_size, hipStream_t stream) {
  const float*         query = (const float*)d_in[0];
  const float*         key   = (const float*)d_in[1];
  const float*         value = (const float*)d_in[2];
  const unsigned char* kpm   = (const unsigned char*)d_in[3];
  const float* Wq = (const float*)d_in[4];  const float* bq = (const float*)d_in[5];
  const float* Wk = (const float*)d_in[6];  const float* bk = (const float*)d_in[7];
  const float* Wv = (const float*)d_in[8];  const float* bv = (const float*)d_in[9];
  const float* Wo = (const float*)d_in[10]; const float* bo = (const float*)d_in[11];
  const float* rb = (const float*)d_in[12];

  // workspace: 4 transposed bf16 weights + q/k(/v transposed) + attn + bias tab
  char* ws = (char*)d_ws;
  size_t off = 0;
  const size_t wsz = (size_t)EMBED * EMBED * 2;
  const size_t hsz = (size_t)BATCH * HEADS * TLEN * HDIM * 2;
  unsigned short* wqb = (unsigned short*)(ws + off); off += wsz;
  unsigned short* wkb = (unsigned short*)(ws + off); off += wsz;
  unsigned short* wvb = (unsigned short*)(ws + off); off += wsz;
  unsigned short* wob = (unsigned short*)(ws + off); off += wsz;
  unsigned short* qhb = (unsigned short*)(ws + off); off += hsz;
  unsigned short* khb = (unsigned short*)(ws + off); off += hsz;
  unsigned short* vhb = (unsigned short*)(ws + off); off += hsz;  // [B,H,D,T]
  unsigned short* atb = (unsigned short*)(ws + off); off += hsz;
  float*          btb = (float*)(ws + off); off += (size_t)HEADS * TBIAS * 4;
  (void)ws_size; (void)in_sizes; (void)n_in; (void)out_size;

  const int nw = EMBED * EMBED;
  k_wcvt_t<<<(nw + 255) / 256, 256, 0, stream>>>(Wq, wqb);
  k_wcvt_t<<<(nw + 255) / 256, 256, 0, stream>>>(Wk, wkb);
  k_wcvt_t<<<(nw + 255) / 256, 256, 0, stream>>>(Wv, wvb);
  k_wcvt_t<<<(nw + 255) / 256, 256, 0, stream>>>(Wo, wob);
  k_biasrel<<<(HEADS * TBIAS + 255) / 256, 256, 0, stream>>>(rb, btb);

  dim3 g1(64, 6, 3);   // 4096/64 M-blocks, 768/128 N-blocks, q/k/v
  k_qkv_gemm<<<g1, 256, 0, stream>>>(query, key, value, wqb, wkb, wvb, bq, bk,
                                     bv, qhb, khb, vhb);

  dim3 g2(TLEN / 64, BATCH * HEADS);
  k_attn<<<g2, 128, 0, stream>>>(qhb, khb, vhb, kpm, btb, atb);

  dim3 g3(64, 6);
  k_out_gemm<<<g3, 256, 0, stream>>>(atb, wob, bo, (float*)d_out);
}
